// DeformableEncoder_51685636440862
// MI455X (gfx1250) — matrix operand order repaired and checked by reading, not compiled, not run on hardware
//
#include <hip/hip_runtime.h>
#include <cstdint>
#include <cstddef>

// ---------------------------------------------------------------------------
// MI455X (gfx1250) deformable-DETR encoder.
// GEMMs run on v_wmma_f32_16x16x32_bf16 (f32 accumulate). A-tiles are staged
// global->LDS with GLOBAL_LOAD_ASYNC_TO_LDS_B128 (ASYNCcnt path); B-tiles are
// K-pair interleaved through VGPRs. MSDA gather is a wave-per-(token,head)
// kernel with lane==channel so corner gathers coalesce (value fits in L2).
// ---------------------------------------------------------------------------

typedef __bf16 bf16_t;
typedef __attribute__((ext_vector_type(16))) __bf16 v16bf;
typedef __attribute__((ext_vector_type(8)))  float  v8f;

#define E_DIM   256
#define T_TOK   13294
#define B_SZ    2
#define MT      (B_SZ * T_TOK)   // 26588 rows
#define NH_     8
#define HD_     32
#define L_      4
#define NP_     4
#define FFN_    1024
#define NLAYERS_ 6

union FragA { v16bf v; uint4 q[2]; };
union FragB { v16bf v; unsigned int u[8]; };
union AccU  { v8f  v; float f[8]; };

// ---------------- f32 -> bf16 weight conversion ----------------------------
__global__ __launch_bounds__(256)
void cvt_bf16_kernel(const float* __restrict__ in, bf16_t* __restrict__ out, int n) {
  int i = blockIdx.x * 256 + threadIdx.x;
  if (i < n) out[i] = (bf16_t)in[i];
}

// ---------------- WMMA bf16 GEMM: C[MxN] = A[MxK] * B[KxN] (+bias/relu/resid)
// Block: 256 threads (8 waves). Block tile 64(M) x 128(N). Wave tile 16x64.
template<bool RELU, bool RESID, bool OUT_BF16>
__global__ __launch_bounds__(256)
void gemm_bf16_wmma(const bf16_t* __restrict__ A, const bf16_t* __restrict__ Bw,
                    const float* __restrict__ bias, const float* __restrict__ resid,
                    void* __restrict__ outp, int M, int N, int K)
{
  // A tile 64x32 bf16, row stride 40 bf16 (80B: 16B aligned for b128 frag reads)
  __shared__ alignas(16) bf16_t       lA[64 * 40];
  // B tile 32x128 packed as K-pairs: lB[p][n] = (B[2p][n], B[2p+1][n]); stride 130
  __shared__ alignas(16) unsigned int lB[16 * 130];

  const int tid  = threadIdx.x;
  const int wave = tid >> 5;
  const int lane = tid & 31;
  const int hv   = lane >> 4;      // half-wave select (K interleave)
  const int lr   = lane & 15;
  const int wm   = wave >> 1;      // 0..3 : M sub-tile
  const int wn   = wave & 1;       // 0..1 : N sub-tile (64 wide)
  const int mblk = blockIdx.x * 64;
  const int nblk = blockIdx.y * 128;

  AccU acc[4];
#pragma unroll
  for (int t = 0; t < 4; ++t)
#pragma unroll
    for (int j = 0; j < 8; ++j) acc[t].f[j] = 0.0f;

  // cooperative load coordinates
  const int arow = tid >> 2;             // 0..63
  const int acol = (tid & 3) * 8;        // bf16 col {0,8,16,24}
  int ag = mblk + arow; if (ag > M - 1) ag = M - 1;   // clamp padding rows
  const int bp  = tid >> 4;              // 0..15 : K-pair index
  const int bnc = (tid & 15) * 8;        // col chunk

  // raw LDS byte offset for the async DMA destination (aperture tag is in the
  // high 32 bits of the generic address; low 32 bits are the LDS address)
  const unsigned int lAoff =
      (unsigned int)(uintptr_t)(const void*)lA + (unsigned int)(arow * 80 + acol * 2);

  for (int k0 = 0; k0 < K; k0 += 32) {
    __syncthreads();
    { // A tile 64x32 bf16: async DMA global -> LDS, 16B per lane (ASYNCcnt)
      const unsigned long long ga =
          (unsigned long long)(uintptr_t)(A + (size_t)ag * K + (k0 + acol));
      asm volatile("global_load_async_to_lds_b128 %0, %1, off"
                   :: "v"(lAoff), "v"(ga) : "memory");
    }
    { // B tile: load rows 2p,2p+1 and interleave into K-pair dwords
      const bf16_t* b0 = Bw + (size_t)(k0 + 2 * bp) * N + (nblk + bnc);
      uint4 r0 = *(const uint4*)b0;
      uint4 r1 = *(const uint4*)(b0 + N);
      unsigned int* dst = lB + bp * 130 + bnc;
      unsigned int lo, hi;
      lo = r0.x; hi = r1.x; dst[0] = (lo & 0xffffu) | (hi << 16); dst[1] = (lo >> 16) | (hi & 0xffff0000u);
      lo = r0.y; hi = r1.y; dst[2] = (lo & 0xffffu) | (hi << 16); dst[3] = (lo >> 16) | (hi & 0xffff0000u);
      lo = r0.z; hi = r1.z; dst[4] = (lo & 0xffffu) | (hi << 16); dst[5] = (lo >> 16) | (hi & 0xffff0000u);
      lo = r0.w; hi = r1.w; dst[6] = (lo & 0xffffu) | (hi << 16); dst[7] = (lo >> 16) | (hi & 0xffff0000u);
    }
    // drain this wave's async copies before publishing the tile at the barrier
    // (__syncthreads only waits DScnt, so the explicit async wait is required)
    asm volatile("s_wait_asynccnt 0x0" ::: "memory");
    __syncthreads();

    if (k0 + 32 < K) {  // pull next K tile toward L2 (global_prefetch_b8)
      __builtin_prefetch((const void*)(A + (size_t)ag * K + (k0 + 32 + acol)), 0, 3);
      __builtin_prefetch((const void*)(Bw + (size_t)(k0 + 32 + 2 * bp) * N + (nblk + bnc)), 0, 3);
    }

    // A fragment (ISA 16-bit A layout): lanes 0-15 K=0..7 / 16..23,
    // lanes 16-31 K=8..15 / 24..31, 2 bf16 per VGPR -> two b128 LDS reads.
    FragA af;
    {
      const char* ab = (const char*)lA + ((wm * 16 + lr) * 40 + hv * 8) * 2;
      af.q[0] = *(const uint4*)ab;
      af.q[1] = *(const uint4*)(ab + 32);
    }
#pragma unroll
    for (int nt = 0; nt < 4; ++nt) {
      // B fragment: lane column n, VGPR v holds K-pair p = (v>>2)*8 + hv*4 + (v&3)
      FragB bf;
      const int ncol = wn * 64 + nt * 16 + lr;
#pragma unroll
      for (int v = 0; v < 8; ++v) bf.u[v] = lB[((v >> 2) * 8 + hv * 4 + (v & 3)) * 130 + ncol];
      acc[nt].v = __builtin_amdgcn_wmma_f32_16x16x32_bf16(
          false, af.v, false, bf.v, (short)0, acc[nt].v, false, false);
    }
  }

  // Epilogue per C/D layout: VGPR v -> M = v + 8*hv, N = lane%16
#pragma unroll
  for (int nt = 0; nt < 4; ++nt) {
    const int n = nblk + wn * 64 + nt * 16 + lr;
#pragma unroll
    for (int v = 0; v < 8; ++v) {
      const int m = mblk + wm * 16 + v + 8 * hv;
      if (m < M) {
        float r = acc[nt].f[v] + bias[n];
        if (RELU)  r = r > 0.0f ? r : 0.0f;
        if (RESID) r += resid[(size_t)m * N + n];
        if (OUT_BF16) ((bf16_t*)outp)[(size_t)m * N + n] = (bf16_t)r;
        else          ((float*) outp)[(size_t)m * N + n] = r;
      }
    }
  }
}

// ---------------- LayerNorm (wave per row) + bf16 x and q = x + pos --------
__global__ __launch_bounds__(256)
void ln_kernel(const float* __restrict__ src, const float* __restrict__ g,
               const float* __restrict__ bb, const float* __restrict__ pos,
               bf16_t* __restrict__ xh, bf16_t* __restrict__ qh, int rows)
{
  const int wave = threadIdx.x >> 5;
  const int lane = threadIdx.x & 31;
  const int row  = blockIdx.x * 8 + wave;
  if (row >= rows) return;
  const float* s = src + (size_t)row * E_DIM;
  float v[8];
  float sum = 0.0f;
#pragma unroll
  for (int i = 0; i < 8; ++i) { v[i] = s[lane + 32 * i]; sum += v[i]; }
#pragma unroll
  for (int m = 16; m >= 1; m >>= 1) sum += __shfl_xor(sum, m, 32);
  const float mean = sum * (1.0f / E_DIM);
  float var = 0.0f;
#pragma unroll
  for (int i = 0; i < 8; ++i) { float d = v[i] - mean; var += d * d; }
#pragma unroll
  for (int m = 16; m >= 1; m >>= 1) var += __shfl_xor(var, m, 32);
  const float rstd = rsqrtf(var * (1.0f / E_DIM) + 1e-5f);
#pragma unroll
  for (int i = 0; i < 8; ++i) {
    const int c = lane + 32 * i;
    const float xn = (v[i] - mean) * rstd * g[c] + bb[c];
    xh[(size_t)row * E_DIM + c] = (bf16_t)xn;
    if (qh) qh[(size_t)row * E_DIM + c] = (bf16_t)(xn + pos[(size_t)row * E_DIM + c]);
  }
}

// ---------------- zero masked value rows -----------------------------------
__global__ __launch_bounds__(256)
void mask_value_kernel(float* __restrict__ value, const unsigned char* __restrict__ mask, int n) {
  int i = blockIdx.x * 256 + threadIdx.x;
  if (i < n && mask[i / E_DIM]) value[i] = 0.0f;
}

// ---------------- softmax over 16 (L*NP) per (token,head) ------------------
__global__ __launch_bounds__(256)
void softmax16_kernel(float* __restrict__ attn, int nrows) {
  int i = blockIdx.x * 256 + threadIdx.x;
  if (i >= nrows) return;
  float* a = attn + (size_t)i * 16;
  float mx = a[0];
#pragma unroll
  for (int j = 1; j < 16; ++j) mx = fmaxf(mx, a[j]);
  float e[16]; float s = 0.0f;
#pragma unroll
  for (int j = 0; j < 16; ++j) { e[j] = __expf(a[j] - mx); s += e[j]; }
  const float inv = 1.0f / s;
#pragma unroll
  for (int j = 0; j < 16; ++j) a[j] = e[j] * inv;
}

// ---------------- reference points per token -------------------------------
__global__ __launch_bounds__(256)
void refpts_kernel(const float* __restrict__ vr, float* __restrict__ ref) {
  int bt = blockIdx.x * 256 + threadIdx.x;
  if (bt >= MT) return;
  const int b = bt / T_TOK;
  const int q = bt % T_TOK;
  int lvl, st, Hs, Ws;
  if      (q >= 13125) { lvl = 3; st = 13125; Hs = 13;  Ws = 13;  }
  else if (q >= 12500) { lvl = 2; st = 12500; Hs = 25;  Ws = 25;  }
  else if (q >= 10000) { lvl = 1; st = 10000; Hs = 50;  Ws = 50;  }
  else                 { lvl = 0; st = 0;     Hs = 100; Ws = 100; }
  const int rel = q - st;
  const int iy = rel / Ws, ix = rel % Ws;
  const float rx = (ix + 0.5f) / (vr[(b * L_ + lvl) * 2 + 0] * Ws);
  const float ry = (iy + 0.5f) / (vr[(b * L_ + lvl) * 2 + 1] * Hs);
#pragma unroll
  for (int l = 0; l < L_; ++l) {
    ref[((size_t)bt * L_ + l) * 2 + 0] = rx * vr[(b * L_ + l) * 2 + 0];
    ref[((size_t)bt * L_ + l) * 2 + 1] = ry * vr[(b * L_ + l) * 2 + 1];
  }
}

// ---------------- MSDA: wave per (token, head), lane == channel d ----------
__global__ __launch_bounds__(256)
void msda_kernel(const float* __restrict__ value, const float* __restrict__ off,
                 const float* __restrict__ attn, const float* __restrict__ ref,
                 bf16_t* __restrict__ out)
{
  const int gw = blockIdx.x * 8 + (threadIdx.x >> 5);
  if (gw >= MT * NH_) return;
  const int d  = threadIdx.x & 31;     // HD == 32 == wave32
  const int bt = gw / NH_;
  const int h  = gw % NH_;
  const int b  = bt / T_TOK;
  const size_t obase = ((size_t)bt * NH_ + h) * (L_ * NP_ * 2);
  const size_t abase = ((size_t)bt * NH_ + h) * (L_ * NP_);
  const int Hs[4] = {100, 50, 25, 13};
  const int Ws[4] = {100, 50, 25, 13};
  const int st[4] = {0, 10000, 12500, 13125};
  float acc = 0.0f;
#pragma unroll
  for (int lvl = 0; lvl < 4; ++lvl) {
    const int W = Ws[lvl], H = Hs[lvl];
    const float rx = ref[((size_t)bt * L_ + lvl) * 2 + 0];
    const float ry = ref[((size_t)bt * L_ + lvl) * 2 + 1];
    const size_t vbase = ((size_t)(b * T_TOK + st[lvl]) * NH_ + h) * HD_ + d;
#pragma unroll
    for (int p = 0; p < NP_; ++p) {
      const float ox = off[obase + (lvl * NP_ + p) * 2 + 0];
      const float oy = off[obase + (lvl * NP_ + p) * 2 + 1];
      const float x  = (rx + ox / W) * W - 0.5f;
      const float y  = (ry + oy / H) * H - 0.5f;
      const float fx = floorf(x), fy = floorf(y);
      const int   x0 = (int)fx,  y0 = (int)fy;
      const float wx = x - fx,   wy = y - fy;
      const float aw = attn[abase + lvl * NP_ + p];
      float s = 0.0f;
#pragma unroll
      for (int cy = 0; cy < 2; ++cy)
#pragma unroll
        for (int cx = 0; cx < 2; ++cx) {
          const int ix = x0 + cx, iy = y0 + cy;
          if (ix >= 0 && ix < W && iy >= 0 && iy < H) {
            const float cw = (cx ? wx : 1.0f - wx) * (cy ? wy : 1.0f - wy);
            s += cw * value[vbase + (size_t)(iy * W + ix) * (NH_ * HD_)];
          }
        }
      acc += aw * s;
    }
  }
  out[(size_t)bt * E_DIM + h * HD_ + d] = (bf16_t)acc;
}

// ---------------------------------------------------------------------------
extern "C" void kernel_launch(void* const* d_in, const int* in_sizes, int n_in,
                              void* d_out, int out_size, void* d_ws, size_t ws_size,
                              hipStream_t stream) {
  (void)in_sizes; (void)n_in; (void)out_size; (void)ws_size;
  const float* src  = (const float*)d_in[0];
  const float* pos  = (const float*)d_in[1];
  const float* vr   = (const float*)d_in[2];
  const unsigned char* mask = (const unsigned char*)d_in[5];  // bool mask; all-zero safe either encoding
  const float* Wv   = (const float*)d_in[6];
  const float* bv   = (const float*)d_in[7];
  const float* Woff = (const float*)d_in[8];
  const float* boff = (const float*)d_in[9];
  const float* Wat  = (const float*)d_in[10];
  const float* bat  = (const float*)d_in[11];
  const float* Wou  = (const float*)d_in[12];
  const float* bou  = (const float*)d_in[13];
  const float* n1g  = (const float*)d_in[14];
  const float* n1b  = (const float*)d_in[15];
  const float* n2g  = (const float*)d_in[16];
  const float* n2b  = (const float*)d_in[17];
  const float* W1   = (const float*)d_in[18];
  const float* b1   = (const float*)d_in[19];
  const float* W2   = (const float*)d_in[20];
  const float* b2   = (const float*)d_in[21];
  float* out = (float*)d_out;            // running src buffer (updated in place)

  // workspace carve
  char* ws = (char*)d_ws;
  size_t o = 0;
  auto carve = [&](size_t bytes) -> void* {
    void* p = ws + o;
    o += (bytes + 255) & ~(size_t)255;
    return p;
  };
  bf16_t* xh     = (bf16_t*)carve((size_t)MT * E_DIM * sizeof(bf16_t));
  bf16_t* qh     = (bf16_t*)carve((size_t)MT * E_DIM * sizeof(bf16_t));
  float*  valbuf = (float*) carve((size_t)MT * E_DIM * sizeof(float));
  float*  offbuf = (float*) carve((size_t)MT * (NH_*L_*NP_*2) * sizeof(float));
  float*  attbuf = (float*) carve((size_t)MT * (NH_*L_*NP_) * sizeof(float));
  bf16_t* msdab  = (bf16_t*)carve((size_t)MT * E_DIM * sizeof(bf16_t));
  bf16_t* h1buf  = (bf16_t*)carve((size_t)MT * FFN_ * sizeof(bf16_t));
  float*  refbuf = (float*) carve((size_t)MT * L_ * 2 * sizeof(float));
  bf16_t* Wvh    = (bf16_t*)carve((size_t)NLAYERS_ * E_DIM * E_DIM * sizeof(bf16_t));
  bf16_t* Woffh  = (bf16_t*)carve((size_t)NLAYERS_ * E_DIM * 256 * sizeof(bf16_t));
  bf16_t* Wath   = (bf16_t*)carve((size_t)NLAYERS_ * E_DIM * 128 * sizeof(bf16_t));
  bf16_t* Wouh   = (bf16_t*)carve((size_t)NLAYERS_ * E_DIM * E_DIM * sizeof(bf16_t));
  bf16_t* W1h    = (bf16_t*)carve((size_t)NLAYERS_ * E_DIM * FFN_ * sizeof(bf16_t));
  bf16_t* W2h    = (bf16_t*)carve((size_t)NLAYERS_ * FFN_ * E_DIM * sizeof(bf16_t));

  const dim3 blk(256);
  const int gM = (MT + 63) / 64;                     // 416 M-blocks

  // init: running src, weight conversion, reference points
  hipMemcpyAsync(out, src, (size_t)MT * E_DIM * sizeof(float),
                 hipMemcpyDeviceToDevice, stream);
  auto cvt = [&](const float* fin, bf16_t* fo, int n) {
    cvt_bf16_kernel<<<(n + 255) / 256, blk, 0, stream>>>(fin, fo, n);
  };
  cvt(Wv,   Wvh,   NLAYERS_ * E_DIM * E_DIM);
  cvt(Woff, Woffh, NLAYERS_ * E_DIM * 256);
  cvt(Wat,  Wath,  NLAYERS_ * E_DIM * 128);
  cvt(Wou,  Wouh,  NLAYERS_ * E_DIM * E_DIM);
  cvt(W1,   W1h,   NLAYERS_ * E_DIM * FFN_);
  cvt(W2,   W2h,   NLAYERS_ * FFN_ * E_DIM);
  refpts_kernel<<<(MT + 255) / 256, blk, 0, stream>>>(vr, refbuf);

  const int lnBlocks = (MT + 7) / 8;
  for (int l = 0; l < NLAYERS_; ++l) {
    const bf16_t* Wvl  = Wvh   + (size_t)l * E_DIM * E_DIM;
    const bf16_t* Wofl = Woffh + (size_t)l * E_DIM * 256;
    const bf16_t* Watl = Wath  + (size_t)l * E_DIM * 128;
    const bf16_t* Woul = Wouh  + (size_t)l * E_DIM * E_DIM;
    const bf16_t* W1l  = W1h   + (size_t)l * E_DIM * FFN_;
    const bf16_t* W2l  = W2h   + (size_t)l * FFN_ * E_DIM;

    // LN1 -> x (bf16), q = x + pos (bf16)
    ln_kernel<<<lnBlocks, blk, 0, stream>>>(out, n1g + l * E_DIM, n1b + l * E_DIM,
                                            pos, xh, qh, MT);
    // value = x @ Wv + bv  (f32)
    gemm_bf16_wmma<false,false,false><<<dim3(gM, 2), blk, 0, stream>>>(
        xh, Wvl, bv + l * E_DIM, nullptr, valbuf, MT, E_DIM, E_DIM);
    mask_value_kernel<<<(MT * E_DIM + 255) / 256, blk, 0, stream>>>(valbuf, mask, MT * E_DIM);
    // off = q @ Woff + boff  (f32)
    gemm_bf16_wmma<false,false,false><<<dim3(gM, 2), blk, 0, stream>>>(
        qh, Wofl, boff + l * 256, nullptr, offbuf, MT, 256, E_DIM);
    // attn logits = q @ Wattn + battn ; softmax over 16
    gemm_bf16_wmma<false,false,false><<<dim3(gM, 1), blk, 0, stream>>>(
        qh, Watl, bat + l * 128, nullptr, attbuf, MT, 128, E_DIM);
    softmax16_kernel<<<(MT * NH_ + 255) / 256, blk, 0, stream>>>(attbuf, MT * NH_);
    // deformable sampling -> bf16
    msda_kernel<<<(MT * NH_ + 7) / 8, blk, 0, stream>>>(valbuf, offbuf, attbuf, refbuf, msdab);
    // src += msda @ Wout + bout
    gemm_bf16_wmma<false,true,false><<<dim3(gM, 2), blk, 0, stream>>>(
        msdab, Woul, bou + l * E_DIM, out, out, MT, E_DIM, E_DIM);
    // LN2 -> x (bf16)
    ln_kernel<<<lnBlocks, blk, 0, stream>>>(out, n2g + l * E_DIM, n2b + l * E_DIM,
                                            nullptr, xh, nullptr, MT);
    // h1 = relu(x @ W1 + b1)  (bf16)
    gemm_bf16_wmma<true,false,true><<<dim3(gM, 8), blk, 0, stream>>>(
        xh, W1l, b1 + l * FFN_, nullptr, h1buf, MT, FFN_, E_DIM);
    // src += h1 @ W2 + b2
    gemm_bf16_wmma<false,true,false><<<dim3(gM, 2), blk, 0, stream>>>(
        h1buf, W2l, b2 + l * E_DIM, out, out, MT, E_DIM, FFN_);
  }
}
